// SlidingWindowAttention_81432579932228
// MI455X (gfx1250) — matrix operand order repaired and checked by reading, not compiled, not run on hardware
//
#include <hip/hip_runtime.h>
#include <hip/hip_bf16.h>
#include <math.h>

// ---------------------------------------------------------------------------
// Problem constants (match reference)
// ---------------------------------------------------------------------------
#define BQ    2      // batch
#define NSEQ  2048   // sequence length
#define CM    1024   // d_model
#define HH    16     // heads
#define DH    64     // head dim
#define HALFW 256    // window/2

// GEMM tiling: 8 waves, wave grid 2x4, each wave 2x2 tiles of 16x16
#define BM 64
#define BN 128
#define BK 32

typedef __attribute__((ext_vector_type(16))) __bf16 v16bf;
typedef __attribute__((ext_vector_type(8)))  __bf16 v8bf;
typedef __attribute__((ext_vector_type(8)))  float  v8f;
typedef __attribute__((ext_vector_type(4)))  unsigned int u32x4;
typedef __attribute__((ext_vector_type(8)))  int    i32x8;
typedef __attribute__((ext_vector_type(4)))  int    i32x4;

// ---------------------------------------------------------------------------
// TDM (Tensor Data Mover) descriptor per CDNA5 ISA ch.8.
// 2D tile: LDS[row*tile_d0 + col] = GLOBAL[start + row*stride0 + col] (bf16).
// ---------------------------------------------------------------------------
#if defined(__has_builtin)
#  if __has_builtin(__builtin_amdgcn_tensor_load_to_lds)
#    define USE_TDM 1
#  endif
#endif
#ifndef USE_TDM
#  define USE_TDM 0
#endif

#if USE_TDM
__device__ __forceinline__ void tdm_load_tile_2d(unsigned lds_byte_addr,
                                                 const void* gaddr,
                                                 unsigned tensor_d0,
                                                 unsigned tensor_d1,
                                                 unsigned tile_d0,
                                                 unsigned tile_d1,
                                                 unsigned long long stride0_elems) {
  unsigned long long ga = (unsigned long long)gaddr;
  u32x4 g0;
  g0.x = 1u;                                               // count=1, user mode
  g0.y = lds_byte_addr;                                    // LDS byte address
  g0.z = (unsigned)(ga & 0xFFFFFFFFu);                     // global_addr[31:0]
  g0.w = (unsigned)((ga >> 32) & 0x1FFFFFFu) | (2u << 30); // addr[56:32] | type=2
  i32x8 g1;
  g1[0] = (int)(1u << 16);                                 // wg_mask=0, data_size=1 (2B)
  g1[1] = (int)((tensor_d0 & 0xFFFFu) << 16);              // tensor_dim0[15:0]
  g1[2] = (int)((tensor_d0 >> 16) | ((tensor_d1 & 0xFFFFu) << 16));
  g1[3] = (int)((tensor_d1 >> 16) | ((tile_d0 & 0xFFFFu) << 16));
  g1[4] = (int)(tile_d1 & 0xFFFFu);                        // tile_dim1, tile_dim2=0
  g1[5] = (int)(stride0_elems & 0xFFFFFFFFull);            // stride0 lo
  g1[6] = (int)((stride0_elems >> 32) & 0xFFFFu);          // stride0 hi, stride1=0
  g1[7] = 0;
  i32x4 z4 = {0, 0, 0, 0};
#if defined(__clang_major__) && __clang_major__ >= 23
  i32x8 z8 = {0, 0, 0, 0, 0, 0, 0, 0};
  __builtin_amdgcn_tensor_load_to_lds(g0, g1, z4, z4, z8, 0);
#else
  __builtin_amdgcn_tensor_load_to_lds(g0, g1, z4, z4, 0);
#endif
}
#endif

// ---------------------------------------------------------------------------
// WMMA helpers. Fragment maps per ISA 7.12.2 (wave32):
//  A 16x32: lane L -> row m=L&15, kb=8*(L>=16); runs k=[kb,kb+8) and [kb+16,kb+24)
//  B 32x16: lane L -> col n=L&15, kb=8*(L>=16); runs k=[kb,kb+8) and [kb+16,kb+24)
//  C 16x16: (vgpr r, lane) -> row r+8*(L>=16), col L&15
// ---------------------------------------------------------------------------
__device__ __forceinline__ v8f wmma_bf16(v16bf a, v16bf b, v8f c) {
  return __builtin_amdgcn_wmma_f32_16x16x32_bf16(false, a, false, b,
                                                 (short)0, c, false, false);
}

__device__ __forceinline__ v16bf cat8(v8bf lo, v8bf hi) {
  return __builtin_shufflevector(lo, hi, 0, 1, 2, 3, 4, 5, 6, 7,
                                 8, 9, 10, 11, 12, 13, 14, 15);
}

// A-fragment from row-major [rows][stride] buffer (LDS or global), 2x b128.
__device__ __forceinline__ v16bf afrag(const __bf16* sm, int mBase, int stride) {
  const int lane = (int)(threadIdx.x & 31);
  const int m  = mBase + (lane & 15);
  const int kb = ((lane >> 4) << 3);               // 0 or 8
  const v8bf* p = (const v8bf*)(sm + (size_t)m * stride + kb);
  return cat8(p[0], p[2]);                         // k: kb..kb+7, kb+16..kb+23
}

// B-fragment from K-contiguous (transposed) [cols][BK] buffer, 2x b128.
__device__ __forceinline__ v16bf bfragT(const __bf16* smt, int nBase, int stride) {
  const int lane = (int)(threadIdx.x & 31);
  const int n  = nBase + (lane & 15);
  const int kb = ((lane >> 4) << 3);               // 0 or 8
  const v8bf* p = (const v8bf*)(smt + (size_t)n * stride + kb);
  return cat8(p[0], p[2]);                         // k: kb..kb+7, kb+16..kb+23
}

// ---------------------------------------------------------------------------
// Kernel 1a: fp32 -> bf16 (straight)
// ---------------------------------------------------------------------------
__global__ __launch_bounds__(256) void cvt_f32_to_bf16(const float* __restrict__ src,
                                                       __bf16* __restrict__ dst, int n) {
  int i = blockIdx.x * 256 + threadIdx.x;
  if (i < n) dst[i] = (__bf16)src[i];
}

// Kernel 1b: fp32 [rows][cols] -> bf16 transposed [cols][rows]
__global__ __launch_bounds__(256) void cvt_transpose_bf16(const float* __restrict__ src,
                                                          __bf16* __restrict__ dst,
                                                          int rows, int cols) {
  int i = blockIdx.x * 256 + threadIdx.x;
  if (i < rows * cols) {
    const int n = i / rows;      // output row  (= src col)
    const int k = i % rows;      // output col  (= src row)
    dst[i] = (__bf16)src[(size_t)k * cols + n];
  }
}

// ---------------------------------------------------------------------------
// Kernel 2/4: bf16 GEMM  C[M,Nfull] = A[M,K] * B[K,Nfull], B given transposed
// Bt[Nfull,K]. TDM double-buffered LDS staging; each wave: 2x2 16x16 tiles.
//   mode 0: scatter bf16 Q/K [B,H,N,DH] and V transposed [B,H,DH,N]
//   mode 1: fp32 out + bias
// ---------------------------------------------------------------------------
__global__ __launch_bounds__(256)
void gemm_bf16_wmma(const __bf16* __restrict__ A, const __bf16* __restrict__ Bt,
                    int M, int Nfull, int K, int mode,
                    __bf16* __restrict__ qw, __bf16* __restrict__ kw,
                    __bf16* __restrict__ vw,
                    float* __restrict__ outp, const float* __restrict__ bias) {
  __shared__ __align__(16) __bf16 smA[2][BM * BK];
  __shared__ __align__(16) __bf16 smB[2][BN * BK];

  const int bm   = blockIdx.x;
  const int bn   = blockIdx.y;
  const int wave = (int)(threadIdx.x >> 5);
  const int lane = (int)(threadIdx.x & 31);
  const int tn0  = (wave & 3) * 2;     // two column tiles
  const int tm0  = (wave >> 2) * 2;    // two row tiles

  v8f c00 = {}, c01 = {}, c10 = {}, c11 = {};

#if USE_TDM
  if (wave == 0) {
    tdm_load_tile_2d((unsigned)(size_t)&smA[0][0],
                     A + (size_t)(bm * BM) * K, (unsigned)K, (unsigned)M,
                     BK, BM, (unsigned long long)K);
    tdm_load_tile_2d((unsigned)(size_t)&smB[0][0],
                     Bt + (size_t)(bn * BN) * K, (unsigned)K, (unsigned)Nfull,
                     BK, BN, (unsigned long long)K);
  }
  int cur = 0;
  for (int k0 = 0; k0 < K; k0 += BK) {
    if (wave == 0) __builtin_amdgcn_s_wait_tensorcnt(0);
    __syncthreads();
    if (wave == 0 && (k0 + BK) < K) {   // overlap next DMA with compute
      tdm_load_tile_2d((unsigned)(size_t)&smA[cur ^ 1][0],
                       A + (size_t)(bm * BM) * K + (k0 + BK),
                       (unsigned)K, (unsigned)M, BK, BM, (unsigned long long)K);
      tdm_load_tile_2d((unsigned)(size_t)&smB[cur ^ 1][0],
                       Bt + (size_t)(bn * BN) * K + (k0 + BK),
                       (unsigned)K, (unsigned)Nfull, BK, BN, (unsigned long long)K);
    }
    v16bf a0 = afrag(&smA[cur][0], (tm0 + 0) * 16, BK);
    v16bf a1 = afrag(&smA[cur][0], (tm0 + 1) * 16, BK);
    v16bf b0 = bfragT(&smB[cur][0], (tn0 + 0) * 16, BK);
    v16bf b1 = bfragT(&smB[cur][0], (tn0 + 1) * 16, BK);
    c00 = wmma_bf16(a0, b0, c00);
    c01 = wmma_bf16(a0, b1, c01);
    c10 = wmma_bf16(a1, b0, c10);
    c11 = wmma_bf16(a1, b1, c11);
    __syncthreads();
    cur ^= 1;
  }
#else
  for (int k0 = 0; k0 < K; k0 += BK) {
    for (int i = (int)threadIdx.x; i < BM * BK; i += 256)
      smA[0][i] = A[(size_t)(bm * BM + i / BK) * K + k0 + (i % BK)];
    for (int i = (int)threadIdx.x; i < BN * BK; i += 256)
      smB[0][i] = Bt[(size_t)(bn * BN + i / BK) * K + k0 + (i % BK)];
    __syncthreads();
    v16bf a0 = afrag(&smA[0][0], (tm0 + 0) * 16, BK);
    v16bf a1 = afrag(&smA[0][0], (tm0 + 1) * 16, BK);
    v16bf b0 = bfragT(&smB[0][0], (tn0 + 0) * 16, BK);
    v16bf b1 = bfragT(&smB[0][0], (tn0 + 1) * 16, BK);
    c00 = wmma_bf16(a0, b0, c00);
    c01 = wmma_bf16(a0, b1, c01);
    c10 = wmma_bf16(a1, b0, c10);
    c11 = wmma_bf16(a1, b1, c11);
    __syncthreads();
  }
#endif

  // Epilogue. C layout: (vgpr r, lane) -> row r+8*(lane>=16), col lane&15.
  const int half = lane >> 4;
  const int nloc = lane & 15;
  v8f cacc[2][2] = {{c00, c01}, {c10, c11}};

#pragma unroll
  for (int jt = 0; jt < 2; ++jt) {
    const int gn = bn * BN + (tn0 + jt) * 16 + nloc;
    if (mode == 0) {
      const int t  = gn >> 10;           // 0=Q 1=K 2=V
      const int rm = gn & (CM - 1);
      const int hh = rm >> 6;
      const int dd = rm & (DH - 1);
      __bf16* dst = (t == 0) ? qw : (t == 1) ? kw : vw;
#pragma unroll
      for (int it = 0; it < 2; ++it) {
#pragma unroll
        for (int r = 0; r < 8; ++r) {
          const int gm = bm * BM + (tm0 + it) * 16 + r + 8 * half;
          const int bb = gm >> 11, nn = gm & (NSEQ - 1);
          const size_t idx = (t == 2)
              ? ((((size_t)bb * HH + hh) * DH + dd) * NSEQ + nn)   // V: [B,H,DH,N]
              : ((((size_t)bb * HH + hh) * NSEQ + nn) * DH + dd);  // Q/K: [B,H,N,DH]
          dst[idx] = (__bf16)cacc[it][jt][r];
        }
      }
    } else {
      const float bv = bias ? bias[gn] : 0.0f;
#pragma unroll
      for (int it = 0; it < 2; ++it) {
#pragma unroll
        for (int r = 0; r < 8; ++r) {
          const int gm = bm * BM + (tm0 + it) * 16 + r + 8 * half;
          outp[(size_t)gm * Nfull + gn] = cacc[it][jt][r] + bv;
        }
      }
    }
  }
}

// ---------------------------------------------------------------------------
// Kernel 3: banded flash attention, bf16 WMMA, fp32 online softmax.
// One wave per 16 q-rows; 8 waves/block => 128 q-rows/block.
// Q/K in [B,H,N,DH]; V in [B,H,DH,N] (transposed) so all fragments are
// contiguous 128-bit loads. Key loop steps by 32.
// ---------------------------------------------------------------------------
__global__ __launch_bounds__(256)
void attn_swa_wmma(const __bf16* __restrict__ Q, const __bf16* __restrict__ Km,
                   const __bf16* __restrict__ Vt, __bf16* __restrict__ ctx) {
  __shared__ __align__(16) __bf16 pbuf[8][16 * 32];  // per-wave P staging

  const int nblk = NSEQ / 128;
  const int qblk = (int)blockIdx.x % nblk;
  const int bh   = (int)blockIdx.x / nblk;
  const int h    = bh % HH;
  const int b    = bh / HH;
  const int wave = (int)(threadIdx.x >> 5);
  const int lane = (int)(threadIdx.x & 31);
  const int q0   = qblk * 128 + wave * 16;

  const __bf16* Qp  = Q  + ((size_t)b * HH + h) * NSEQ * DH;
  const __bf16* Kp  = Km + ((size_t)b * HH + h) * NSEQ * DH;
  const __bf16* Vtp = Vt + ((size_t)b * HH + h) * (size_t)DH * NSEQ;

  const float scale = 0.125f;  // DH^-0.5
  const int half = lane >> 4;
  const int nloc = lane & 15;
  const int kbA  = ((lane >> 4) << 3);   // A-frag K base: 0/8
  const int kbB  = ((lane >> 4) << 3);   // B-frag K base: 0/8

  // Q fragments for d=[0,32) and [32,64): contiguous 128-bit loads
  v16bf qf0, qf1;
  {
    const int m = lane & 15;
    const v8bf* p0 = (const v8bf*)(Qp + (size_t)(q0 + m) * DH + kbA);
    const v8bf* p1 = (const v8bf*)(Qp + (size_t)(q0 + m) * DH + 32 + kbA);
    qf0 = cat8(p0[0], p0[2]);
    qf1 = cat8(p1[0], p1[2]);
  }

  v8f mrow, lrow;
  v8f acc0 = {}, acc1 = {}, acc2 = {}, acc3 = {};
#pragma unroll
  for (int r = 0; r < 8; ++r) { mrow[r] = -INFINITY; lrow[r] = 0.0f; }

  int js = q0 - HALFW; if (js < 0) js = 0; js &= ~31;
  int je = q0 + 15 + HALFW; if (je > NSEQ - 1) je = NSEQ - 1;
  // N%32==0 and js 32-aligned => all visited j in [0, NSEQ): no bounds guards.

  __bf16* pb = pbuf[wave];

  for (int j0 = js; j0 <= je; j0 += 32) {
    if (j0 + 32 <= je) {  // prefetch next key/value tiles
      __builtin_prefetch((const void*)(Kp + (size_t)(j0 + 32) * DH), 0, 1);
      __builtin_prefetch((const void*)(Vtp + (size_t)nloc * NSEQ + j0 + 32), 0, 1);
    }

    // ---- S = scale * Q K^T for keys [j0, j0+32) ----
    v8f s0 = {}, s1 = {};
#pragma unroll
    for (int kh = 0; kh < 2; ++kh) {
      v8f sacc = {};
#pragma unroll
      for (int dh = 0; dh < 2; ++dh) {
        const int j = j0 + kh * 16 + nloc;
        const v8bf* kp = (const v8bf*)(Kp + (size_t)j * DH + dh * 32 + kbB);
        v16bf kf = cat8(kp[0], kp[2]);
        sacc = wmma_bf16(dh ? qf1 : qf0, kf, sacc);
      }
      if (kh) s1 = sacc; else s0 = sacc;
    }

    // ---- band mask + scale ----
#pragma unroll
    for (int r = 0; r < 8; ++r) {
      const int i = q0 + r + 8 * half;
      int da = i - (j0 + nloc);      if (da < 0) da = -da;
      int db = i - (j0 + 16 + nloc); if (db < 0) db = -db;
      s0[r] = (da <= HALFW) ? s0[r] * scale : -INFINITY;
      s1[r] = (db <= HALFW) ? s1[r] * scale : -INFINITY;
    }

    // ---- online softmax: row reductions across the 16 lanes of each half ----
    v8f tmax;
#pragma unroll
    for (int r = 0; r < 8; ++r) tmax[r] = fmaxf(s0[r], s1[r]);
#pragma unroll
    for (int off = 1; off < 16; off <<= 1) {
#pragma unroll
      for (int r = 0; r < 8; ++r)
        tmax[r] = fmaxf(tmax[r], __shfl_xor(tmax[r], off, 32));
    }

    v8f alpha;
#pragma unroll
    for (int r = 0; r < 8; ++r) {
      const float mn = fmaxf(mrow[r], tmax[r]);
      alpha[r] = (mn == -INFINITY) ? 1.0f : __expf(mrow[r] - mn);
      mrow[r]  = mn;
      s0[r] = (s0[r] == -INFINITY) ? 0.0f : __expf(s0[r] - mn);
      s1[r] = (s1[r] == -INFINITY) ? 0.0f : __expf(s1[r] - mn);
    }

    v8f rsum;
#pragma unroll
    for (int r = 0; r < 8; ++r) rsum[r] = s0[r] + s1[r];
#pragma unroll
    for (int off = 1; off < 16; off <<= 1) {
#pragma unroll
      for (int r = 0; r < 8; ++r)
        rsum[r] += __shfl_xor(rsum[r], off, 32);
    }
#pragma unroll
    for (int r = 0; r < 8; ++r) lrow[r] = lrow[r] * alpha[r] + rsum[r];

#pragma unroll
    for (int r = 0; r < 8; ++r) {
      acc0[r] *= alpha[r]; acc1[r] *= alpha[r];
      acc2[r] *= alpha[r]; acc3[r] *= alpha[r];
    }

    // ---- P: C-layout fp32 -> LDS bf16 -> A-fragment (b128 reads) ----
#pragma unroll
    for (int r = 0; r < 8; ++r) {
      const int m = r + 8 * half;
      pb[m * 32 + nloc]      = (__bf16)s0[r];
      pb[m * 32 + 16 + nloc] = (__bf16)s1[r];
    }
    __builtin_amdgcn_wave_barrier();
    v16bf pf;
    {
      const int m = lane & 15;
      const v8bf* pp = (const v8bf*)(pb + m * 32 + kbA);
      pf = cat8(pp[0], pp[2]);
    }

    // ---- O += P V  (V transposed: key index is contiguous) ----
#pragma unroll
    for (int t = 0; t < 4; ++t) {
      const int dcol = t * 16 + nloc;
      const v8bf* vp = (const v8bf*)(Vtp + (size_t)dcol * NSEQ + j0 + kbB);
      v16bf vf = cat8(vp[0], vp[2]);
      v8f a = (t == 0) ? acc0 : (t == 1) ? acc1 : (t == 2) ? acc2 : acc3;
      a = wmma_bf16(pf, vf, a);
      if (t == 0) acc0 = a; else if (t == 1) acc1 = a;
      else if (t == 2) acc2 = a; else acc3 = a;
    }
  }

  // ---- finalize: out = acc / l, store bf16 context [B,N,C] ----
#pragma unroll
  for (int t = 0; t < 4; ++t) {
    v8f a = (t == 0) ? acc0 : (t == 1) ? acc1 : (t == 2) ? acc2 : acc3;
#pragma unroll
    for (int r = 0; r < 8; ++r) {
      const float inv = (lrow[r] > 0.0f) ? (1.0f / lrow[r]) : 0.0f;
      const int m = r + 8 * half;
      const int d = t * 16 + nloc;
      ctx[((size_t)(b * NSEQ + q0 + m)) * CM + h * DH + d] = (__bf16)(a[r] * inv);
    }
  }
}

// ---------------------------------------------------------------------------
// Host launcher
// ---------------------------------------------------------------------------
extern "C" void kernel_launch(void* const* d_in, const int* in_sizes, int n_in,
                              void* d_out, int out_size, void* d_ws, size_t ws_size,
                              hipStream_t stream) {
  (void)in_sizes; (void)n_in; (void)out_size; (void)ws_size;

  const float* x     = (const float*)d_in[0];
  const float* Wqkv  = (const float*)d_in[1];
  const float* Wproj = (const float*)d_in[2];
  const float* bproj = (const float*)d_in[3];
  float* out = (float*)d_out;

  const int M  = BQ * NSEQ;  // 4096
  const int K  = CM;         // 1024
  const int N3 = 3 * CM;     // 3072
  const size_t nHeadTensor = (size_t)BQ * HH * NSEQ * DH;

  __bf16* xb   = (__bf16*)d_ws;            // [M,K]
  __bf16* wqbT = xb   + (size_t)M * K;     // [N3,K]  (W_qkv transposed)
  __bf16* wpbT = wqbT + (size_t)N3 * K;    // [CM,K]  (W_proj transposed)
  __bf16* qws  = wpbT + (size_t)CM * K;    // [B,H,N,DH]
  __bf16* kws  = qws + nHeadTensor;        // [B,H,N,DH]
  __bf16* vws  = kws + nHeadTensor;        // [B,H,DH,N] (transposed)
  __bf16* ctxb = vws + nHeadTensor;        // [M,CM]

  { const int n = M * K;
    cvt_f32_to_bf16<<<(n + 255) / 256, 256, 0, stream>>>(x, xb, n); }
  { const int n = K * N3;
    cvt_transpose_bf16<<<(n + 255) / 256, 256, 0, stream>>>(Wqkv, wqbT, K, N3); }
  { const int n = K * CM;
    cvt_transpose_bf16<<<(n + 255) / 256, 256, 0, stream>>>(Wproj, wpbT, K, CM); }

  // QKV projection: [4096,1024] x [1024,3072] -> Q/K [B,H,N,DH], V [B,H,DH,N]
  gemm_bf16_wmma<<<dim3(M / BM, N3 / BN), 256, 0, stream>>>(
      xb, wqbT, M, N3, K, /*mode=*/0, qws, kws, vws, nullptr, nullptr);

  // Banded attention
  attn_swa_wmma<<<dim3(BQ * HH * (NSEQ / 128)), 256, 0, stream>>>(qws, kws, vws, ctxb);

  // Output projection + bias: [4096,1024] x [1024,1024] -> fp32 out
  gemm_bf16_wmma<<<dim3(M / BM, CM / BN), 256, 0, stream>>>(
      ctxb, wpbT, M, CM, K, /*mode=*/1, nullptr, nullptr, nullptr, out, bproj);
}
